// GraphConv_12927851561318
// MI455X (gfx1250) — matrix operand order repaired and checked by reading, not compiled, not run on hardware
//
#include <hip/hip_runtime.h>
#include <hip/hip_bf16.h>

// ---------------------------------------------------------------------------
// Types for CDNA5 WMMA
// ---------------------------------------------------------------------------
typedef __bf16 v16bf __attribute__((ext_vector_type(16)));
typedef float  v8f   __attribute__((ext_vector_type(8)));

__device__ __forceinline__ unsigned short f32_to_bf16_bits(float f) {
    union { float f; unsigned int u; } x; x.f = f;
    unsigned int u = x.u;
    u += 0x7FFFu + ((u >> 16) & 1u);   // round-to-nearest-even
    return (unsigned short)(u >> 16);
}

// ordered-uint encoding of -inf (monotone float->uint map)
#define ORDERED_NEG_INF 0x007FFFFFu

// ---------------------------------------------------------------------------
// WMMA bf16 GEMM:  C[M,N] = alpha * A[M,K] @ op(B)
//   A row-major (lda); transB==0: B is KxN (ldb); transB==1: B is NxK (ldb)
//
// Block: 256 threads = 8 waves. Block tile 128(M) x 64(N) x 32(K).
// Waves arranged 4(M) x 2(N); each wave owns a 32x32 output tile =
// 2x2 v_wmma_f32_16x16x32_bf16 tiles (A/B fragments reused twice each).
//
// LDS holds A/B tiles PRE-SWIZZLED into per-lane WMMA fragment order so a
// fragment load is two ds_load_b128 per matrix (not 16 ds_load_u16):
//   A 16x32 sub-tile:  lane = m + 16*((k>>3)&1)
//                      sidx = 2*((k>>4)*4 + ((k&7)>>1)) + (k&1)
//   B 32x16 sub-tile:  lane = n + 16*((k>>3)&1),  sidx = 8*(k>>4) + (k&7)
// ---------------------------------------------------------------------------
#define BM 128
#define BN 64
#define BK 32

__global__ __launch_bounds__(256)
void gemm_bf16_wmma(const float* __restrict__ A, const float* __restrict__ B,
                    float* __restrict__ C, int M, int N, int K,
                    int lda, int ldb, int ldc, int transB, float alpha)
{
    // [subtile][lane][16 bf16] ; each lane's fragment is 32 contiguous bytes
    __shared__ __align__(32) unsigned short Aswz[BM / 16][32][16];
    __shared__ __align__(32) unsigned short Bswz[BN / 16][32][16];

    const int tid  = threadIdx.x;
    const int lane = tid & 31;
    const int wave = tid >> 5;      // 0..7
    const int wm   = wave >> 1;     // 0..3  (32-row band)
    const int wn   = wave & 1;      // 0..1  (32-col band)
    const int bm   = blockIdx.y * BM;
    const int bn   = blockIdx.x * BN;

    v8f acc00 = {}, acc01 = {}, acc10 = {}, acc11 = {};

    for (int k0 = 0; k0 < K; k0 += BK) {
        // ---- stage A tile (128x32) into swizzled bf16 LDS
        for (int t = tid; t < BM * BK; t += 256) {
            int r = t >> 5;              // 0..127
            int c = t & 31;              // 0..31 (k within tile)
            int gr = bm + r, gc = k0 + c;
            float v = (gr < M && gc < K) ? A[(size_t)gr * lda + gc] : 0.0f;
            int sub  = r >> 4;
            int m    = r & 15;
            int ln   = m + 16 * ((c >> 3) & 1);
            int sidx = 2 * ((c >> 4) * 4 + ((c & 7) >> 1)) + (c & 1);
            Aswz[sub][ln][sidx] = f32_to_bf16_bits(v);
        }
        // ---- stage B tile (32x64) into swizzled bf16 LDS
        for (int t = tid; t < BK * BN; t += 256) {
            int k = t >> 6;              // 0..31
            int c = t & 63;              // 0..63 (n within tile)
            int gk = k0 + k, gn = bn + c;
            float v = 0.0f;
            if (gk < K && gn < N)
                v = transB ? B[(size_t)gn * ldb + gk]
                           : B[(size_t)gk * ldb + gn];
            int sub  = c >> 4;
            int n    = c & 15;
            int ln   = n + 16 * ((k >> 3) & 1);
            Bswz[sub][ln][8 * (k >> 4) + (k & 7)] = f32_to_bf16_bits(v);
        }
        __syncthreads();

        // ---- fragment loads: 2x ds_load_b128 each
        v16bf a0 = *(const v16bf*)&Aswz[wm * 2 + 0][lane][0];
        v16bf a1 = *(const v16bf*)&Aswz[wm * 2 + 1][lane][0];
        v16bf b0 = *(const v16bf*)&Bswz[wn * 2 + 0][lane][0];
        v16bf b1 = *(const v16bf*)&Bswz[wn * 2 + 1][lane][0];

        acc00 = __builtin_amdgcn_wmma_f32_16x16x32_bf16(
                    false, a0, false, b0, (short)0, acc00, false, false);
        acc01 = __builtin_amdgcn_wmma_f32_16x16x32_bf16(
                    false, a0, false, b1, (short)0, acc01, false, false);
        acc10 = __builtin_amdgcn_wmma_f32_16x16x32_bf16(
                    false, a1, false, b0, (short)0, acc10, false, false);
        acc11 = __builtin_amdgcn_wmma_f32_16x16x32_bf16(
                    false, a1, false, b1, (short)0, acc11, false, false);
        __syncthreads();
    }

    // ---- store: C/D layout row = r + 8*(lane>>4), col = lane&15
    const int l15  = lane & 15;
    const int lh8  = 8 * (lane >> 4);
    #pragma unroll
    for (int i = 0; i < 2; ++i) {
        #pragma unroll
        for (int j = 0; j < 2; ++j) {
            const v8f* accp = (i == 0) ? (j == 0 ? &acc00 : &acc01)
                                       : (j == 0 ? &acc10 : &acc11);
            #pragma unroll
            for (int r = 0; r < 8; ++r) {
                int row = bm + (wm * 2 + i) * 16 + r + lh8;
                int col = bn + (wn * 2 + j) * 16 + l15;
                if (row < M && col < N)
                    C[(size_t)row * ldc + col] = alpha * (*accp)[r];
            }
        }
    }
}

// ---------------------------------------------------------------------------
// BatchNorm (training-mode) column stats + affine + LeakyReLU
// ---------------------------------------------------------------------------
__global__ void bn_stats_kernel(const float* __restrict__ y, int M, int N,
                                float* __restrict__ mean, float* __restrict__ rstd)
{
    __shared__ float ssum[256];
    __shared__ float ssq[256];
    int c = blockIdx.x;
    float s = 0.0f, q = 0.0f;
    for (int r = threadIdx.x; r < M; r += blockDim.x) {
        float v = y[(size_t)r * N + c];
        s += v; q += v * v;
    }
    ssum[threadIdx.x] = s; ssq[threadIdx.x] = q;
    __syncthreads();
    for (int st = 128; st > 0; st >>= 1) {
        if (threadIdx.x < st) {
            ssum[threadIdx.x] += ssum[threadIdx.x + st];
            ssq[threadIdx.x]  += ssq[threadIdx.x + st];
        }
        __syncthreads();
    }
    if (threadIdx.x == 0) {
        float m = ssum[0] / (float)M;
        float var = ssq[0] / (float)M - m * m;
        mean[c] = m;
        rstd[c] = rsqrtf(var + 1e-5f);
    }
}

__global__ void bn_act_kernel(float* __restrict__ y, long long total, int N,
                              const float* __restrict__ mean,
                              const float* __restrict__ rstd,
                              const float* __restrict__ g,
                              const float* __restrict__ b, float slope)
{
    long long i = (long long)blockIdx.x * blockDim.x + threadIdx.x;
    if (i >= total) return;
    int c = (int)(i % N);
    float v = (y[i] - mean[c]) * rstd[c] * g[c] + b[c];
    y[i] = v > 0.0f ? v : slope * v;
}

// ---------------------------------------------------------------------------
// Static graph conv: scatter-max of edge messages via ordered-uint atomicMax
// msg cols: [ x[dst] (d) | x[src]-x[dst] (d) | edge_attr (64) ]
// ---------------------------------------------------------------------------
__device__ __forceinline__ unsigned int float_to_ordered(float f) {
    union { float f; unsigned int u; } x; x.f = f;
    return (x.u & 0x80000000u) ? ~x.u : (x.u | 0x80000000u);
}
__device__ __forceinline__ float ordered_to_float(unsigned int u) {
    union { float f; unsigned int u; } x;
    x.u = (u & 0x80000000u) ? (u ^ 0x80000000u) : ~u;
    return x.f;
}

__global__ void fill_u32_kernel(unsigned int* __restrict__ p, unsigned int v,
                                long long n)
{
    long long i = (long long)blockIdx.x * blockDim.x + threadIdx.x;
    if (i < n) p[i] = v;
}

__global__ void edge_scatter_max_kernel(const float* __restrict__ x,
                                        const float* __restrict__ ea,
                                        const int* __restrict__ src,
                                        const int* __restrict__ dst,
                                        unsigned int* __restrict__ agg,
                                        long long total, int d, int cols)
{
    long long t = (long long)blockIdx.x * blockDim.x + threadIdx.x;
    if (t >= total) return;
    int c = (int)(t % cols);
    long long e = t / cols;
    int di = dst[e];
    float v;
    if (c < d) {
        v = x[(size_t)di * d + c];
    } else if (c < 2 * d) {
        int si = src[e];
        int cc = c - d;
        v = x[(size_t)si * d + cc] - x[(size_t)di * d + cc];
    } else {
        v = ea[e * 64 + (c - 2 * d)];
    }
    atomicMax(&agg[(size_t)di * cols + c], float_to_ordered(v));
}

__global__ void agg_decode_kernel(float* __restrict__ agg, long long n)
{
    long long i = (long long)blockIdx.x * blockDim.x + threadIdx.x;
    if (i >= n) return;
    unsigned int u = ((unsigned int*)agg)[i];
    float f = ordered_to_float(u);
    if (!isfinite(f)) f = 0.0f;   // isolated nodes -> 0 (PyG max-aggr fill)
    agg[i] = f;
}

// ---------------------------------------------------------------------------
// Dynamic conv helpers: squared norms, top-16 NN scan, edge features, row-max
// ---------------------------------------------------------------------------
__global__ void sqnorm_kernel(const float* __restrict__ x, float* __restrict__ sq,
                              int n, int d)
{
    int i = blockIdx.x * blockDim.x + threadIdx.x;
    if (i >= n) return;
    float s = 0.0f;
    for (int c = 0; c < d; ++c) {
        float v = x[(size_t)i * d + c];
        s += v * v;
    }
    sq[i] = s;
}

__global__ void knn16_kernel(const float* __restrict__ G,
                             const float* __restrict__ sq,
                             int* __restrict__ idx, int n)
{
    int i = blockIdx.x * blockDim.x + threadIdx.x;
    if (i >= n) return;
    float bd[16];
    int   bi[16];
    #pragma unroll
    for (int k = 0; k < 16; ++k) { bd[k] = __builtin_inff(); bi[k] = 0; }
    const float* gi = G + (size_t)i * n;
    float si = sq[i];
    for (int j = 0; j < n; ++j) {
        __builtin_prefetch(gi + j + 128, 0, 1);
        if (j == i) continue;
        float dist = si + sq[j] - 2.0f * gi[j];
        if (dist < bd[15]) {
            int p = 15;
            while (p > 0 && bd[p - 1] > dist) {
                bd[p] = bd[p - 1]; bi[p] = bi[p - 1]; --p;
            }
            bd[p] = dist; bi[p] = j;
        }
    }
    #pragma unroll
    for (int k = 0; k < 16; ++k) idx[i * 16 + k] = bi[k];
}

// msg[(i*16+k)] = [ x[idx]-x[i] (d) | x[i] (d) ]
__global__ void dyn_msg_kernel(const float* __restrict__ x,
                               const int* __restrict__ idx,
                               float* __restrict__ msg, long long total, int d)
{
    long long t = (long long)blockIdx.x * blockDim.x + threadIdx.x;
    if (t >= total) return;
    int twoD = 2 * d;
    int c = (int)(t % twoD);
    long long row = t / twoD;
    int i = (int)(row >> 4);
    int k = (int)(row & 15);
    int j = idx[i * 16 + k];
    float v;
    if (c < d) v = x[(size_t)j * d + c] - x[(size_t)i * d + c];
    else       v = x[(size_t)i * d + (c - d)];
    msg[t] = v;
}

__global__ void rowmax16_kernel(const float* __restrict__ h,
                                float* __restrict__ out, long long total, int d)
{
    long long t = (long long)blockIdx.x * blockDim.x + threadIdx.x;
    if (t >= total) return;
    int c = (int)(t % d);
    long long i = t / d;
    float m = -__builtin_inff();
    #pragma unroll
    for (int k = 0; k < 16; ++k)
        m = fmaxf(m, h[((size_t)i * 16 + k) * d + c]);
    out[t] = m;
}

// ---------------------------------------------------------------------------
// Column-slice copy for concatenation
// ---------------------------------------------------------------------------
__global__ void copy_cols_kernel(const float* __restrict__ src,
                                 float* __restrict__ dst, long long total,
                                 int sN, int dN, int off)
{
    long long t = (long long)blockIdx.x * blockDim.x + threadIdx.x;
    if (t >= total) return;
    int c = (int)(t % sN);
    long long r = t / sN;
    dst[r * dN + off + c] = src[t];
}

// ---------------------------------------------------------------------------
// Host orchestration
// ---------------------------------------------------------------------------
static inline unsigned int blocks_for(long long n) {
    return (unsigned int)((n + 255) / 256);
}

extern "C" void kernel_launch(void* const* d_in, const int* in_sizes, int n_in,
                              void* d_out, int out_size, void* d_ws, size_t ws_size,
                              hipStream_t stream)
{
    (void)in_sizes; (void)out_size; (void)ws_size;

    const int NN = 10000;
    const int E  = 320000;

    const float* x_in = (const float*)d_in[0];
    const float* ea   = (const float*)d_in[1];
    const int* e_src  = (const int*)d_in[n_in - 2];
    const int* e_dst  = (const int*)d_in[n_in - 1];

    // Flattened params, setup_inputs() insertion order:
    // sg1(9) sg2(9) sg3(9) dg1(6) dg2(6) fuse1(3) fuse2(6) fuse2_out(1)
    const float* prm[64];
    int np = 0;
    for (int i = 2; i < n_in - 2 && np < 64; ++i)
        prm[np++] = (const float*)d_in[i];
    int pc = 0;

    // ---- workspace bump allocator (256B aligned)
    char* wsb = (char*)d_ws;
    size_t off = 0;
    auto alloc = [&](size_t nfloats) -> float* {
        off = (off + 255) & ~(size_t)255;
        float* p = (float*)(wsb + off);
        off += nfloats * sizeof(float);
        return p;
    };

    float* G    = alloc(100000000);          // 10000x10000 Gram (400 MB)
    float* msgA = G;                          // reuse after kNN idx extracted
    float* msgB = G + 41000000;               // 160000x256 fits (40.96M)
    float* s1   = alloc((size_t)NN * 64);
    float* s2   = alloc((size_t)NN * 128);
    float* s3   = alloc((size_t)NN * 192);
    float* d1   = alloc((size_t)NN * 128);
    float* d2   = alloc((size_t)NN * 192);
    float* agg  = alloc((size_t)NN * 320);
    float* tA   = alloc((size_t)NN * 512);
    float* tB   = alloc((size_t)NN * 256);
    float* cat1 = alloc((size_t)NN * 704);
    float* f1   = alloc((size_t)NN * 1024);
    float* cat2 = alloc((size_t)NN * 1728);
    float* sq   = alloc(NN);
    int*   idx  = (int*)alloc(NN * 16);
    float* mean = alloc(2048);
    float* rstd = alloc(2048);

    // ---- helpers
    auto gemm = [&](const float* A, const float* B, float* C,
                    int M, int N, int K, int transB, float alpha) {
        dim3 grid((N + BN - 1) / BN, (M + BM - 1) / BM);
        gemm_bf16_wmma<<<grid, dim3(256), 0, stream>>>(
            A, B, C, M, N, K, K, transB ? K : N, N, transB, alpha);
    };
    auto bnact = [&](float* Y, int M, int N, const float* g, const float* b,
                     float slope) {
        bn_stats_kernel<<<dim3(N), dim3(256), 0, stream>>>(Y, M, N, mean, rstd);
        long long tot = (long long)M * N;
        bn_act_kernel<<<blocks_for(tot), 256, 0, stream>>>(Y, tot, N, mean, rstd,
                                                           g, b, slope);
    };
    auto mlp_layer = [&](const float* X, float* Y, int M, int K, int N,
                         float slope) {
        const float* w = prm[pc++];
        const float* g = prm[pc++];
        const float* b = prm[pc++];
        gemm(X, w, Y, M, N, K, 0, 1.0f);
        bnact(Y, M, N, g, b, slope);
    };

    // ---- static conv: agg = segment_max(msg, dst) then MLP chain
    auto sconv = [&](const float* xin, int d, const int* Ldims, int nl,
                     float* out) {
        int cols = 2 * d + 64;
        long long na = (long long)NN * cols;
        fill_u32_kernel<<<blocks_for(na), 256, 0, stream>>>(
            (unsigned int*)agg, ORDERED_NEG_INF, na);
        long long te = (long long)E * cols;
        edge_scatter_max_kernel<<<blocks_for(te), 256, 0, stream>>>(
            xin, ea, e_src, e_dst, (unsigned int*)agg, te, d, cols);
        agg_decode_kernel<<<blocks_for(na), 256, 0, stream>>>(agg, na);
        const float* cur = agg;
        int curd = cols;
        float* bufs[2] = {tA, tB};
        for (int l = 0; l < nl; ++l) {
            float* Y = (l == nl - 1) ? out : bufs[l & 1];
            mlp_layer(cur, Y, NN, curd, Ldims[l], 0.1f);
            cur = Y; curd = Ldims[l];
        }
    };

    // ---- dynamic conv: knn(16) + EdgeConv MLP + max over K
    auto dconv = [&](const float* xin, int d, const int* Ldims, int nl,
                     float* out) {
        sqnorm_kernel<<<blocks_for(NN), 256, 0, stream>>>(xin, sq, NN, d);
        gemm(xin, xin, G, NN, NN, d, /*transB=*/1, 1.0f);      // Gram = x @ x^T
        knn16_kernel<<<blocks_for(NN), 256, 0, stream>>>(G, sq, idx, NN);
        long long rows = (long long)NN * 16;
        long long tm = rows * 2 * d;
        dyn_msg_kernel<<<blocks_for(tm), 256, 0, stream>>>(xin, idx, msgA, tm, d);
        const float* cur = msgA;
        int curd = 2 * d;
        for (int l = 0; l < nl; ++l) {
            float* Y = (l & 1) ? msgA : msgB;
            mlp_layer(cur, Y, (int)rows, curd, Ldims[l], 0.2f);
            cur = Y; curd = Ldims[l];
        }
        long long to = (long long)NN * curd;
        rowmax16_kernel<<<blocks_for(to), 256, 0, stream>>>(cur, out, to, curd);
    };

    // ---- forward pass
    static const int L1[3] = {128, 64, 64};
    static const int L2[3] = {160, 128, 128};
    static const int L3[3] = {256, 192, 192};
    static const int D1[2] = {128, 128};
    static const int D2[2] = {192, 192};

    sconv(x_in, 64, L1, 3, s1);   // sg1 -> [n,64]
    sconv(s1, 64, L2, 3, s2);     // sg2 -> [n,128]
    sconv(s2, 128, L3, 3, s3);    // sg3 -> [n,192]
    dconv(s1, 64, D1, 2, d1);     // dg1 -> [n,128]
    dconv(d1, 128, D2, 2, d2);    // dg2 -> [n,192]

    // cat1 = [s1, d1, d2, s2, s3]  -> [n,704]
    auto cpy = [&](const float* s, int sN, float* d, int dN, int o) {
        long long t = (long long)NN * sN;
        copy_cols_kernel<<<blocks_for(t), 256, 0, stream>>>(s, d, t, sN, dN, o);
    };
    cpy(s1, 64, cat1, 704, 0);
    cpy(d1, 128, cat1, 704, 64);
    cpy(d2, 192, cat1, 704, 192);
    cpy(s2, 128, cat1, 704, 384);
    cpy(s3, 192, cat1, 704, 512);

    mlp_layer(cat1, f1, NN, 704, 1024, 0.2f);   // fuse1

    // cat2 = [f1, cat1] -> [n,1728]
    cpy(f1, 1024, cat2, 1728, 0);
    cpy(cat1, 704, cat2, 1728, 1024);

    mlp_layer(cat2, tA, NN, 1728, 512, 0.2f);   // fuse2[0]
    mlp_layer(tA, tB, NN, 512, 256, 0.2f);      // fuse2[1]

    // final projection (no BN/act): out = tB @ fuse2_out  -> [n,128]
    const float* wout = prm[pc++];
    gemm(tB, wout, (float*)d_out, NN, 128, 256, 0, 1.0f);
}
